// MPNN_63367947485958
// MI455X (gfx1250) — matrix operand order repaired and checked by reading, not compiled, not run on hardware
//
#include <hip/hip_runtime.h>

#define NN 100000   // nodes
#define NE 800000   // edges
#define NG 5000     // graphs

typedef __attribute__((ext_vector_type(16))) _Float16 v16h;
typedef __attribute__((ext_vector_type(8)))  float    v8f;

__device__ __forceinline__ v8f wmma16(v16h a, v16h b, v8f c) {
  // D = A(16x32 f16) * B(32x16 f16) + C(16x16 f32)
  return __builtin_amdgcn_wmma_f32_16x16x32_f16(false, a, false, b, (short)0, c,
                                                false, false);
}

// ---------------------------------------------------------------------------
// Node-side kernels (tiny FLOP count -> plain VALU)
// ---------------------------------------------------------------------------
__global__ void k_init1(const float* __restrict__ x, const float* __restrict__ root,
                        const float* __restrict__ bias, float* __restrict__ tmp1) {
  int t = blockIdx.x * blockDim.x + threadIdx.x;
  if (t >= NN * 32) return;
  int n = t >> 5, c = t & 31;
  float s = bias[c];
  #pragma unroll
  for (int i = 0; i < 11; ++i) s += x[n * 11 + i] * root[i * 32 + c];
  tmp1[t] = s;
}

__global__ void k_relu1(const float* __restrict__ tmp1, float* __restrict__ h1r) {
  int t = blockIdx.x * blockDim.x + threadIdx.x;
  if (t >= NN * 32) return;
  h1r[t] = fmaxf(tmp1[t], 0.f);
}

__global__ void k_init2(const float* __restrict__ h1r, const float* __restrict__ root,
                        const float* __restrict__ bias, float* __restrict__ tmp2) {
  int t = blockIdx.x * blockDim.x + threadIdx.x;
  if (t >= NN * 16) return;
  int n = t >> 4, c = t & 15;
  float s = bias[c];
  #pragma unroll
  for (int k = 0; k < 32; ++k) s += h1r[n * 32 + k] * root[k * 16 + c];
  tmp2[t] = s;
}

__global__ void k_zero(float* __restrict__ pooled) {
  int t = blockIdx.x * blockDim.x + threadIdx.x;
  if (t < NG * 16) pooled[t] = 0.f;
}

__global__ void k_pool(const float* __restrict__ tmp2, const int* __restrict__ batch,
                       float* __restrict__ pooled) {
  int t = blockIdx.x * blockDim.x + threadIdx.x;
  if (t >= NN * 16) return;
  int n = t >> 4, c = t & 15;
  atomicAdd(&pooled[batch[n] * 16 + c], fmaxf(tmp2[t], 0.f));
}

__global__ void k_head(const float* __restrict__ pooled,
                       const float* __restrict__ fc1w, const float* __restrict__ fc1b,
                       const float* __restrict__ outw, const float* __restrict__ outb,
                       float* __restrict__ out) {
  int g = blockIdx.x * blockDim.x + threadIdx.x;
  if (g >= NG) return;
  float p[16];
  #pragma unroll
  for (int c = 0; c < 16; ++c) p[c] = pooled[g * 16 + c];
  float o = outb[0];
  #pragma unroll 4
  for (int c2 = 0; c2 < 32; ++c2) {
    float s = fc1b[c2];
    #pragma unroll
    for (int c = 0; c < 16; ++c) s += p[c] * fc1w[c * 32 + c2];
    o += fmaxf(s, 0.f) * outw[c2];
  }
  out[g] = o;
}

// ---------------------------------------------------------------------------
// conv1 fused edge kernel.
//   Per wave, 16 edges = WMMA M dimension.
//   A[16 x 384] = outer-product features f[e,(i,k)] = x_src[e,i] * hidden[e,k]
//                 (11 K=32 chunks) + bias chunk (f = x_src zero-padded).
//   B[384 x 32] = c1_w2 reshaped (+ c1_b2 chunk), staged in LDS pre-swizzled
//                 into WMMA B-fragment order so each fragment is ONE contiguous
//                 32-byte per-lane read (2x ds_load_b128, conflict-free).
//   D[16 x 32] f32 -> scatter-add via global_atomic_add_f32.
// ---------------------------------------------------------------------------
__global__ __launch_bounds__(128) void k_edge1(
    const float* __restrict__ x, const int* __restrict__ ei,
    const float* __restrict__ eattr,
    const float* __restrict__ w1, const float* __restrict__ b1,
    const float* __restrict__ w2, const float* __restrict__ b2,
    float* __restrict__ tmp1) {
  // [chunk(12)][tile(2)][lane(32)][j(16)] f16, fragment order
  __shared__ __attribute__((aligned(32))) _Float16 bw[12 * 2 * 32 * 16];
  for (int t = threadIdx.x; t < 12 * 2 * 32 * 16; t += 128) {
    const int j    = t & 15;
    const int lane = (t >> 4) & 31;
    const int rest = t >> 9;
    const int ot   = rest & 1;
    const int i    = rest >> 1;
    const int kb   = (j < 8) ? ((lane >> 4) * 8 + j) : (16 + (lane >> 4) * 8 + (j - 8));      // B layout: table order, matches A
    const int o    = ot * 16 + (lane & 15);
    float v;
    if (i < 11) v = w2[kb * 352 + i * 32 + o];
    else        v = (kb < 11) ? b2[kb * 32 + o] : 0.f;   // bias chunk, zero pad
    bw[t] = (_Float16)v;
  }
  __syncthreads();

  const int wave = threadIdx.x >> 5;
  const int lane = threadIdx.x & 31;
  const int l16  = lane & 15;
  const int hi   = lane >> 4;
  const int* __restrict__ srcp = ei;
  const int* __restrict__ dstp = ei + NE;

  for (int g = 0; g < 4; ++g) {
    const int ebase = blockIdx.x * 256 + wave * 64 + g * 16;
    const int e   = ebase + l16;
    const int src = srcp[e];

    const float4 ea = ((const float4*)eattr)[e];

    // Edge-MLP hidden units at this lane's A-fragment K positions, packed f16:
    // hi==0 -> k in {0..7,16..23}; hi==1 -> k in {8..15,24..31}
    v16h hkh;
    #pragma unroll
    for (int j = 0; j < 16; ++j) {
      const int k = (j < 8) ? (hi * 8 + j) : (16 + hi * 8 + (j - 8));
      float s = b1[k];
      s += ea.x * w1[0 * 32 + k];
      s += ea.y * w1[1 * 32 + k];
      s += ea.z * w1[2 * 32 + k];
      s += ea.w * w1[3 * 32 + k];
      hkh[j] = (_Float16)fmaxf(s, 0.f);
    }

    _Float16 xh[11];
    #pragma unroll
    for (int i = 0; i < 11; ++i) xh[i] = (_Float16)x[src * 11 + i];

    v8f acc0 = {}; v8f acc1 = {};

    #pragma unroll
    for (int i = 0; i < 11; ++i) {              // K chunks: (i, k=0..31)
      const v16h af  = hkh * xh[i];             // 8x v_pk_mul_f16
      const v16h bf0 = *(const v16h*)(bw + ((i * 2 + 0) * 32 + lane) * 16);
      const v16h bf1 = *(const v16h*)(bw + ((i * 2 + 1) * 32 + lane) * 16);
      acc0 = wmma16(af, bf0, acc0);
      acc1 = wmma16(af, bf1, acc1);
    }
    {   // bias chunk: A[k'] = (k'<11) ? x[k'] : 0
      v16h af;
      const _Float16 z = (_Float16)0.f;
      #pragma unroll
      for (int j = 0; j < 16; ++j) {
        const _Float16 v0 = (j < 8) ? xh[j] : z;       // hi==0: k'=j else >=16
        const _Float16 v1 = (j < 3) ? xh[8 + j] : z;   // hi==1: k'=8+j valid iff <11
        af[j] = hi ? v1 : v0;
      }
      const v16h bf0 = *(const v16h*)(bw + ((11 * 2 + 0) * 32 + lane) * 16);
      const v16h bf1 = *(const v16h*)(bw + ((11 * 2 + 1) * 32 + lane) * 16);
      acc0 = wmma16(af, bf0, acc0);
      acc1 = wmma16(af, bf1, acc1);
    }

    // D layout: VGPR r <-> row M = r + 8*hi, col N = l16 (+16 for acc1)
    #pragma unroll
    for (int r = 0; r < 8; ++r) {
      const int em = ebase + r + 8 * hi;
      const int d  = dstp[em];
      atomicAdd(&tmp1[d * 32 + l16],      acc0[r]);
      atomicAdd(&tmp1[d * 32 + 16 + l16], acc1[r]);
    }
  }
}

// ---------------------------------------------------------------------------
// conv2 fused edge kernel: c_in=32, c_out=16 -> K = 32*32 + 32 = 1056 (33 chunks),
// one 16-wide N tile. Same structure as k_edge1.
// ---------------------------------------------------------------------------
__global__ __launch_bounds__(128) void k_edge2(
    const float* __restrict__ h1r, const int* __restrict__ ei,
    const float* __restrict__ eattr,
    const float* __restrict__ w1, const float* __restrict__ b1,
    const float* __restrict__ w2, const float* __restrict__ b2,
    float* __restrict__ tmp2) {
  // [chunk(33)][lane(32)][j(16)] f16, fragment order
  __shared__ __attribute__((aligned(32))) _Float16 bw[33 * 32 * 16];
  for (int t = threadIdx.x; t < 33 * 32 * 16; t += 128) {
    const int j    = t & 15;
    const int lane = (t >> 4) & 31;
    const int i    = t >> 9;
    const int kb   = (j < 8) ? ((lane >> 4) * 8 + j) : (16 + (lane >> 4) * 8 + (j - 8));
    const int o    = lane & 15;
    float v;
    if (i < 32) v = w2[kb * 512 + i * 16 + o];
    else        v = b2[kb * 16 + o];             // bias chunk (all 32 k valid)
    bw[t] = (_Float16)v;
  }
  __syncthreads();

  const int wave = threadIdx.x >> 5;
  const int lane = threadIdx.x & 31;
  const int l16  = lane & 15;
  const int hi   = lane >> 4;
  const int* __restrict__ srcp = ei;
  const int* __restrict__ dstp = ei + NE;

  for (int g = 0; g < 4; ++g) {
    const int ebase = blockIdx.x * 256 + wave * 64 + g * 16;
    const int e   = ebase + l16;
    const int src = srcp[e];

    const float4 ea = ((const float4*)eattr)[e];

    v16h hkh;
    #pragma unroll
    for (int j = 0; j < 16; ++j) {
      const int k = (j < 8) ? (hi * 8 + j) : (16 + hi * 8 + (j - 8));
      float s = b1[k];
      s += ea.x * w1[0 * 32 + k];
      s += ea.y * w1[1 * 32 + k];
      s += ea.z * w1[2 * 32 + k];
      s += ea.w * w1[3 * 32 + k];
      hkh[j] = (_Float16)fmaxf(s, 0.f);
    }

    _Float16 xh[32];
    #pragma unroll
    for (int q = 0; q < 8; ++q) {
      const float4 v = ((const float4*)(h1r + (size_t)src * 32))[q];
      xh[4 * q + 0] = (_Float16)v.x; xh[4 * q + 1] = (_Float16)v.y;
      xh[4 * q + 2] = (_Float16)v.z; xh[4 * q + 3] = (_Float16)v.w;
    }

    v8f acc = {};

    #pragma unroll
    for (int i = 0; i < 32; ++i) {
      const v16h af = hkh * xh[i];
      const v16h bf = *(const v16h*)(bw + (i * 32 + lane) * 16);
      acc = wmma16(af, bf, acc);
    }
    {   // bias chunk: A[k'] = h1r_src[k'] at this lane's A K-positions
      v16h af;
      #pragma unroll
      for (int j = 0; j < 16; ++j) {
        const _Float16 v0 = (j < 8) ? xh[j]     : xh[16 + (j - 8)];  // hi==0
        const _Float16 v1 = (j < 8) ? xh[8 + j] : xh[24 + (j - 8)];  // hi==1
        af[j] = hi ? v1 : v0;
      }
      const v16h bf = *(const v16h*)(bw + (32 * 32 + lane) * 16);
      acc = wmma16(af, bf, acc);
    }

    #pragma unroll
    for (int r = 0; r < 8; ++r) {
      const int em = ebase + r + 8 * hi;
      const int d  = dstp[em];
      atomicAdd(&tmp2[d * 16 + l16], acc[r]);
    }
  }
}

// ---------------------------------------------------------------------------
extern "C" void kernel_launch(void* const* d_in, const int* in_sizes, int n_in,
                              void* d_out, int out_size, void* d_ws, size_t ws_size,
                              hipStream_t stream) {
  (void)in_sizes; (void)n_in; (void)out_size; (void)ws_size;

  const float* x       = (const float*)d_in[0];
  const int*   ei      = (const int*)d_in[1];
  const float* eattr   = (const float*)d_in[2];
  const int*   batch   = (const int*)d_in[3];
  const float* c1_w1   = (const float*)d_in[4];
  const float* c1_b1   = (const float*)d_in[5];
  const float* c1_w2   = (const float*)d_in[6];
  const float* c1_b2   = (const float*)d_in[7];
  const float* c1_root = (const float*)d_in[8];
  const float* c1_bias = (const float*)d_in[9];
  const float* c2_w1   = (const float*)d_in[10];
  const float* c2_b1   = (const float*)d_in[11];
  const float* c2_w2   = (const float*)d_in[12];
  const float* c2_b2   = (const float*)d_in[13];
  const float* c2_root = (const float*)d_in[14];
  const float* c2_bias = (const float*)d_in[15];
  const float* fc1_w   = (const float*)d_in[16];
  const float* fc1_b   = (const float*)d_in[17];
  const float* out_w   = (const float*)d_in[18];
  const float* out_b   = (const float*)d_in[19];
  float* out = (float*)d_out;

  // workspace: tmp1[NN*32] | h1r[NN*32] | tmp2[NN*16] | pooled[NG*16]  (~32.3 MB)
  float* ws     = (float*)d_ws;
  float* tmp1   = ws;
  float* h1r    = ws + (size_t)NN * 32;
  float* tmp2   = ws + (size_t)NN * 64;
  float* pooled = ws + (size_t)NN * 80;

  k_init1<<<(NN * 32 + 255) / 256, 256, 0, stream>>>(x, c1_root, c1_bias, tmp1);
  k_edge1<<<NE / 256, 128, 0, stream>>>(x, ei, eattr, c1_w1, c1_b1, c1_w2, c1_b2, tmp1);
  k_relu1<<<(NN * 32 + 255) / 256, 256, 0, stream>>>(tmp1, h1r);
  k_init2<<<(NN * 16 + 255) / 256, 256, 0, stream>>>(h1r, c2_root, c2_bias, tmp2);
  k_edge2<<<NE / 256, 128, 0, stream>>>(h1r, ei, eattr, c2_w1, c2_b1, c2_w2, c2_b2, tmp2);
  k_zero<<<(NG * 16 + 255) / 256, 256, 0, stream>>>(pooled);
  k_pool<<<(NN * 16 + 255) / 256, 256, 0, stream>>>(tmp2, batch, pooled);
  k_head<<<(NG + 255) / 256, 256, 0, stream>>>(pooled, fc1_w, fc1_b, out_w, out_b, out);
}
